// GVP_embedding_14491219657352
// MI455X (gfx1250) — matrix operand order repaired and checked by reading, not compiled, not run on hardware
//
#include <hip/hip_runtime.h>

#define EPSF 1e-8f
#define NNODE 20000
#define NEDGE 600000

typedef __attribute__((ext_vector_type(16))) __bf16 v16bf;
typedef __attribute__((ext_vector_type(8)))  float v8f;
typedef __attribute__((ext_vector_type(4)))  unsigned int v4u;

union Frag { v16bf v; v4u q[2]; };

__device__ __forceinline__ unsigned short f2bf(float f) {
    unsigned u = __float_as_uint(f);
    unsigned r = u + 0x7FFFu + ((u >> 16) & 1u);
    return (unsigned short)(r >> 16);
}

__device__ __forceinline__ float sigm(float x) { return 1.f / (1.f + expf(-x)); }

// ---------------- WMMA tile GEMM -------------------------------------------
// out[MTILES*16 x NTILES*16] = A[MTILES*16 x KT*32] x W^T
// lA: bf16 LDS rows of ASTRIDE halves. W: pre-swizzled bf16 B fragments.
// 4 waves; wave w owns N-tiles {w, w+4, w+8, ...}; B frag loaded once and
// reused across the MTILES A-tiles (halves weight traffic per row).
template <int KT, int ASTRIDE, int NTILES, int MTILES>
__device__ __forceinline__ void gemm_tile(const unsigned short* lA,
                                          const unsigned short* __restrict__ W,
                                          float* lOut, int lane, int wid) {
    const int m     = lane & 15;
    const int kbase = (lane >> 4) * 16;  // A frag: lanes0-15 K 0-15, lanes16-31 K 16-31
    const int nloc  = lane & 15;
    const int mbase = (lane >> 4) * 8;   // D frag: M = r + 8*(lane/16)
#pragma unroll
    for (int t = 0; t < NTILES / 4; ++t) {
        const int nt = wid + 4 * t;
        v8f acc[MTILES] = {};
#pragma unroll
        for (int kt = 0; kt < KT; ++kt) {
            Frag bf;
            const unsigned short* pb = W + ((size_t)(nt * KT + kt) * 32 + lane) * 16;
            bf.q[0] = *(const v4u*)(pb);
            bf.q[1] = *(const v4u*)(pb + 8);
#pragma unroll
            for (int mt = 0; mt < MTILES; ++mt) {
                Frag af;
                const unsigned short* pa = lA + (mt * 16 + m) * ASTRIDE + kt * 32 + kbase;
                af.q[0] = *(const v4u*)(pa);
                af.q[1] = *(const v4u*)(pa + 8);
                acc[mt] = __builtin_amdgcn_wmma_f32_16x16x32_bf16(
                    false, af.v, false, bf.v, (short)0, acc[mt], false, false);
            }
        }
#pragma unroll
        for (int mt = 0; mt < MTILES; ++mt)
#pragma unroll
            for (int r = 0; r < 8; ++r)
                lOut[(mt * 16 + mbase + r) * (NTILES * 16) + nt * 16 + nloc] = acc[mt][r];
    }
}

// ---------------- prep kernels ---------------------------------------------
__global__ void k_transpose(const float* __restrict__ src, float* __restrict__ dst,
                            int O, int K) {
    int i = blockIdx.x * 256 + threadIdx.x;
    if (i < O * K) { int o = i / K, k = i - o * K; dst[k * O + o] = src[i]; }
}

// ws [NT*16][K] f32 -> bf16 B-fragment order: idx = ((nt*KT+kt)*32 + lane)*16 + h
__global__ void k_swizzle(const float* __restrict__ ws, unsigned short* __restrict__ dst,
                          int K, int KT, int NT) {
    int i = blockIdx.x * 256 + threadIdx.x;
    int total = NT * KT * 512;
    if (i >= total) return;
    int h = i & 15, L = (i >> 4) & 31, t = i >> 9;
    int kt = t % KT, nt = t / KT;
    int n = L & 15, k = kt * 32 + (L >> 4) * 16 + h, o = nt * 16 + n;
    float x = (k < K) ? ws[(size_t)o * K + k] : 0.f;
    dst[i] = f2bf(x);
}

__global__ void k_zero(float* p, int n) {
    int i = blockIdx.x * 256 + threadIdx.x;
    if (i < n) p[i] = 0.f;
}

__global__ void k_count(const int* __restrict__ dst, float* cnt, int E) {
    int i = blockIdx.x * 256 + threadIdx.x;
    if (i < E) atomicAdd(&cnt[dst[i]], 1.f);
}

// ---------------- shared LN / RMS helpers (16 rows x 128, 128 threads) -----
__device__ __forceinline__ void ln16(float* x, float* red, float* st1, float* st2,
                                     const float* g, const float* b, int tid) {
    const int e = tid >> 3, sub = tid & 7;
    float p = 0.f;
    for (int k = sub; k < 128; k += 8) p += x[e * 128 + k];
    red[tid] = p;
    __syncthreads();
    if (sub == 0) {
        float s = 0.f;
        for (int i = 0; i < 8; ++i) s += red[e * 8 + i];
        st1[e] = s * (1.f / 128.f);
    }
    __syncthreads();
    const float mu = st1[e];
    p = 0.f;
    for (int k = sub; k < 128; k += 8) { float d = x[e * 128 + k] - mu; p += d * d; }
    red[tid] = p;
    __syncthreads();
    if (sub == 0) {
        float s = 0.f;
        for (int i = 0; i < 8; ++i) s += red[e * 8 + i];
        st2[e] = rsqrtf(s * (1.f / 128.f) + 1e-5f);
    }
    __syncthreads();
    const float inv = st2[e];
    for (int k = sub; k < 128; k += 8)
        x[e * 128 + k] = (x[e * 128 + k] - mu) * inv * g[k] + b[k];
    __syncthreads();
}

__device__ __forceinline__ void rms16(float* lv, float* st, int tid) {
    if (tid < 16) {
        float s = 0.f;
        for (int ch = 0; ch < 16; ++ch) {
            float a0 = lv[tid * 48 + ch * 3], a1 = lv[tid * 48 + ch * 3 + 1],
                  a2 = lv[tid * 48 + ch * 3 + 2];
            s += fmaxf(a0 * a0 + a1 * a1 + a2 * a2, EPSF);
        }
        st[tid] = rsqrtf(s * (1.f / 16.f));
    }
    __syncthreads();
    for (int i = tid; i < 16 * 48; i += 128) lv[i] *= st[i / 48];
    __syncthreads();
}

// ---------------- node embedding (VALU, tiny) ------------------------------
struct NodeEmbArgs {
    const float *node_s, *node_v; const int* seq;
    const float *Ws, *lng, *lnb, *wh, *wsT, *wv, *b;
    float *S, *V;
};
__launch_bounds__(128)
__global__ void k_node_embed(NodeEmbArgs a) {
    __shared__ float ls0[42], lv[9], lvh[48], lstat[4];
    int n = blockIdx.x, tid = threadIdx.x;
    if (tid < 26) ls0[tid] = tid < 6 ? a.node_s[n * 6 + tid]
                                     : a.Ws[a.seq[n] * 20 + (tid - 6)];
    if (tid < 9) lv[tid] = a.node_v[n * 9 + tid];
    __syncthreads();
    if (tid == 0) {
        float mu = 0.f;
        for (int i = 0; i < 26; ++i) mu += ls0[i];
        mu *= (1.f / 26.f);
        float var = 0.f;
        for (int i = 0; i < 26; ++i) { float d = ls0[i] - mu; var += d * d; }
        var *= (1.f / 26.f);
        lstat[0] = mu; lstat[1] = rsqrtf(var + 1e-5f);
        float s = 0.f;
        for (int ch = 0; ch < 3; ++ch) {
            float n0 = lv[ch * 3], n1 = lv[ch * 3 + 1], n2 = lv[ch * 3 + 2];
            s += fmaxf(n0 * n0 + n1 * n1 + n2 * n2, EPSF);
        }
        lstat[2] = rsqrtf(s * (1.f / 3.f));
    }
    __syncthreads();
    if (tid < 26) ls0[tid] = (ls0[tid] - lstat[0]) * lstat[1] * a.lng[tid] + a.lnb[tid];
    if (tid < 9) lv[tid] *= lstat[2];
    __syncthreads();
    if (tid < 48) {
        int h = tid / 3, c = tid % 3; float s = 0.f;
        for (int vv = 0; vv < 3; ++vv) s += a.wh[h * 3 + vv] * lv[vv * 3 + c];
        lvh[tid] = s;
    }
    __syncthreads();
    if (tid < 16) {
        float n0 = lvh[tid * 3], n1 = lvh[tid * 3 + 1], n2 = lvh[tid * 3 + 2];
        ls0[26 + tid] = sqrtf(fmaxf(n0 * n0 + n1 * n1 + n2 * n2, EPSF));
    }
    __syncthreads();
    float s = a.b[tid];
    for (int k = 0; k < 42; ++k) s += a.wsT[k * 128 + tid] * ls0[k];
    a.S[n * 128 + tid] = s;
    if (tid < 48) {
        int o = tid / 3, c = tid % 3; float t = 0.f;
        for (int h = 0; h < 16; ++h) t += a.wv[o * 16 + h] * lvh[h * 3 + c];
        a.V[n * 48 + tid] = t;
    }
}

// ---------------- edge embedding -------------------------------------------
struct EdgeEmbArgs {
    const float *edge_s, *edge_v, *lng, *lnb, *wh, *ws, *wv, *b;
    float *ES, *EV; int E;
};
__launch_bounds__(256)
__global__ void k_edge_embed(EdgeEmbArgs a) {
    __shared__ float lws[32 * 33], lg[32], lb[32], lbb[32];
    for (int i = threadIdx.x; i < 32 * 33; i += 256) lws[i] = a.ws[i];
    if (threadIdx.x < 32) {
        lg[threadIdx.x] = a.lng[threadIdx.x];
        lb[threadIdx.x] = a.lnb[threadIdx.x];
        lbb[threadIdx.x] = a.b[threadIdx.x];
    }
    __syncthreads();
    int e = blockIdx.x * 256 + threadIdx.x;
    if (e >= a.E) return;
    float x[32]; float mu = 0.f;
#pragma unroll
    for (int k = 0; k < 32; ++k) { x[k] = a.edge_s[(size_t)e * 32 + k]; mu += x[k]; }
    mu *= (1.f / 32.f);
    float var = 0.f;
#pragma unroll
    for (int k = 0; k < 32; ++k) { float d = x[k] - mu; var += d * d; }
    float inv = rsqrtf(var * (1.f / 32.f) + 1e-5f);
#pragma unroll
    for (int k = 0; k < 32; ++k) x[k] = (x[k] - mu) * inv * lg[k] + lb[k];
    float v0 = a.edge_v[e * 3], v1 = a.edge_v[e * 3 + 1], v2 = a.edge_v[e * 3 + 2];
    float vinv = rsqrtf(fmaxf(v0 * v0 + v1 * v1 + v2 * v2, EPSF));
    v0 *= vinv; v1 *= vinv; v2 *= vinv;
    float w = a.wh[0];
    float h0 = w * v0, h1 = w * v1, h2 = w * v2;
    float hn = sqrtf(fmaxf(h0 * h0 + h1 * h1 + h2 * h2, EPSF));
    for (int o = 0; o < 32; ++o) {
        float s = lbb[o];
#pragma unroll
        for (int k = 0; k < 32; ++k) s += lws[o * 33 + k] * x[k];
        s += lws[o * 33 + 32] * hn;
        a.ES[(size_t)e * 32 + o] = s;
    }
    float wo = a.wv[0];
    a.EV[e * 3] = wo * h0; a.EV[e * 3 + 1] = wo * h1; a.EV[e * 3 + 2] = wo * h2;
}

// ---------------- per-edge message kernel (WMMA, 32 edges/block) -----------
struct EdgeArgs {
    const float *S, *V, *ES, *EV;
    const int *src, *dst;
    float *DS, *DV;
    const unsigned short *W0, *W1, *W2;
    const float *b0, *b1, *b2;
    const float *wh0, *wv0, *wh1, *wv1, *wh2, *wv2;
};
__launch_bounds__(128)
__global__ void k_edge_msg(EdgeArgs a) {
    __shared__ alignas(16) unsigned short lA[32 * 352];   // bf16 A staging (K padded)
    __shared__ float lMV[32 * 99];
    __shared__ float lVH[32 * 99];
    __shared__ float lVO[32 * 48];
    __shared__ float lOut[32 * 128];
    __shared__ int lsrc[32], ldst[32];
    const int tid = threadIdx.x, lane = tid & 31, wid = tid >> 5;
    const int e0 = blockIdx.x * 32;

    if (tid < 32) { lsrc[tid] = a.src[e0 + tid]; ldst[tid] = a.dst[e0 + tid]; }
    for (int i = tid; i < 32 * 352; i += 128) lA[i] = 0;
    __syncthreads();

    // gather scalar message: [s_j(128), es(32), s_i(128)] -> bf16
    for (int i = tid; i < 32 * 288; i += 128) {
        int e = i / 288, k = i - e * 288; float x;
        if (k < 128)      x = a.S[(size_t)lsrc[e] * 128 + k];
        else if (k < 160) x = a.ES[(size_t)(e0 + e) * 32 + (k - 128)];
        else              x = a.S[(size_t)ldst[e] * 128 + (k - 160)];
        lA[e * 352 + k] = f2bf(x);
    }
    // gather vector message: [v_j(16), ev(1), v_i(16)] x3
    for (int i = tid; i < 32 * 99; i += 128) {
        int e = i / 99, j = i - e * 99, vv = j / 3, c = j - vv * 3; float x;
        if (vv < 16)       x = a.V[(size_t)lsrc[e] * 48 + vv * 3 + c];
        else if (vv == 16) x = a.EV[(size_t)(e0 + e) * 3 + c];
        else               x = a.V[(size_t)ldst[e] * 48 + (vv - 17) * 3 + c];
        lMV[e * 99 + j] = x;
    }
    __syncthreads();

    // vh0 = wh0[33x33] @ mv ; vn0 -> A cols 288..320
    for (int i = tid; i < 32 * 33; i += 128) {
        int e = i / 33, h = i - e * 33;
        float s0 = 0, s1 = 0, s2 = 0;
        for (int vv = 0; vv < 33; ++vv) {
            float w = a.wh0[h * 33 + vv];
            s0 += w * lMV[e * 99 + vv * 3];
            s1 += w * lMV[e * 99 + vv * 3 + 1];
            s2 += w * lMV[e * 99 + vv * 3 + 2];
        }
        lVH[e * 99 + h * 3] = s0; lVH[e * 99 + h * 3 + 1] = s1; lVH[e * 99 + h * 3 + 2] = s2;
        lA[e * 352 + 288 + h] = f2bf(sqrtf(fmaxf(s0 * s0 + s1 * s1 + s2 * s2, EPSF)));
    }
    __syncthreads();

    gemm_tile<11, 352, 8, 2>(lA, a.W0, lOut, lane, wid);   // msg0 ws: K=321 (pad 352)
    __syncthreads();

    // vo0 = sigmoid-gate(wv0[16x33] @ vh0)
    for (int i = tid; i < 32 * 16; i += 128) {
        int e = i >> 4, o = i & 15;
        float s0 = 0, s1 = 0, s2 = 0;
        for (int h = 0; h < 33; ++h) {
            float w = a.wv0[o * 33 + h];
            s0 += w * lVH[e * 99 + h * 3];
            s1 += w * lVH[e * 99 + h * 3 + 1];
            s2 += w * lVH[e * 99 + h * 3 + 2];
        }
        float g = sigm(sqrtf(fmaxf(s0 * s0 + s1 * s1 + s2 * s2, EPSF)));
        lVO[e * 48 + o * 3] = s0 * g; lVO[e * 48 + o * 3 + 1] = s1 * g; lVO[e * 48 + o * 3 + 2] = s2 * g;
    }
    __syncthreads();

    // vh1 = wh1 @ vo0 ; vn1 -> A cols 128..143 ; s1 = relu(out + b0) -> A cols 0..127
    for (int i = tid; i < 32 * 16; i += 128) {
        int e = i >> 4, h = i & 15;
        float s0 = 0, s1 = 0, s2 = 0;
        for (int vv = 0; vv < 16; ++vv) {
            float w = a.wh1[h * 16 + vv];
            s0 += w * lVO[e * 48 + vv * 3];
            s1 += w * lVO[e * 48 + vv * 3 + 1];
            s2 += w * lVO[e * 48 + vv * 3 + 2];
        }
        lVH[e * 99 + h * 3] = s0; lVH[e * 99 + h * 3 + 1] = s1; lVH[e * 99 + h * 3 + 2] = s2;
        lA[e * 352 + 128 + h] = f2bf(sqrtf(fmaxf(s0 * s0 + s1 * s1 + s2 * s2, EPSF)));
    }
    for (int i = tid; i < 32 * 128; i += 128) {
        int e = i >> 7, k = i & 127;
        float x = lOut[i] + a.b0[k];
        lA[e * 352 + k] = f2bf(fmaxf(x, 0.f));
    }
    for (int i = tid; i < 32 * 16; i += 128) lA[(i >> 4) * 352 + 144 + (i & 15)] = 0;
    __syncthreads();

    gemm_tile<5, 352, 8, 2>(lA, a.W1, lOut, lane, wid);    // msg1 ws: K=144 (pad 160)
    __syncthreads();

    // vo1 = gate(wv1 @ vh1)
    for (int i = tid; i < 32 * 16; i += 128) {
        int e = i >> 4, o = i & 15;
        float s0 = 0, s1 = 0, s2 = 0;
        for (int h = 0; h < 16; ++h) {
            float w = a.wv1[o * 16 + h];
            s0 += w * lVH[e * 99 + h * 3];
            s1 += w * lVH[e * 99 + h * 3 + 1];
            s2 += w * lVH[e * 99 + h * 3 + 2];
        }
        float g = sigm(sqrtf(fmaxf(s0 * s0 + s1 * s1 + s2 * s2, EPSF)));
        lVO[e * 48 + o * 3] = s0 * g; lVO[e * 48 + o * 3 + 1] = s1 * g; lVO[e * 48 + o * 3 + 2] = s2 * g;
    }
    __syncthreads();

    // vh2 = wh2 @ vo1 ; vn2 ; s2 = relu(out + b1)
    for (int i = tid; i < 32 * 16; i += 128) {
        int e = i >> 4, h = i & 15;
        float s0 = 0, s1 = 0, s2 = 0;
        for (int vv = 0; vv < 16; ++vv) {
            float w = a.wh2[h * 16 + vv];
            s0 += w * lVO[e * 48 + vv * 3];
            s1 += w * lVO[e * 48 + vv * 3 + 1];
            s2 += w * lVO[e * 48 + vv * 3 + 2];
        }
        lVH[e * 99 + h * 3] = s0; lVH[e * 99 + h * 3 + 1] = s1; lVH[e * 99 + h * 3 + 2] = s2;
        lA[e * 352 + 128 + h] = f2bf(sqrtf(fmaxf(s0 * s0 + s1 * s1 + s2 * s2, EPSF)));
    }
    for (int i = tid; i < 32 * 128; i += 128) {
        int e = i >> 7, k = i & 127;
        float x = lOut[i] + a.b1[k];
        lA[e * 352 + k] = f2bf(fmaxf(x, 0.f));
    }
    __syncthreads();

    gemm_tile<5, 352, 8, 2>(lA, a.W2, lOut, lane, wid);    // msg2 ws: K=144
    __syncthreads();

    // scatter: dv += wv2 @ vh2 (no gate), ds += out + b2 (no relu)
    for (int i = tid; i < 32 * 48; i += 128) {
        int e = i / 48, j = i - e * 48, o = j / 3, c = j - o * 3;
        float s = 0.f;
        for (int h = 0; h < 16; ++h) s += a.wv2[o * 16 + h] * lVH[e * 99 + h * 3 + c];
        atomicAdd(&a.DV[(size_t)ldst[e] * 48 + j], s);
    }
    for (int i = tid; i < 32 * 128; i += 128) {
        int e = i >> 7, k = i & 127;
        atomicAdd(&a.DS[(size_t)ldst[e] * 128 + k], lOut[i] + a.b2[k]);
    }
}

// ---------------- per-node update (WMMA feed-forward, 16 nodes/block) ------
struct NodeFFArgs {
    float *S, *V;
    const float *DS, *DV, *CNT;
    const float *ln0b, *ln0g, *ln1b, *ln1g;
    const float *ff0b, *ff0wh, *ff0wv;
    const float *ff1b, *ff1wh, *ff1wv;
    const unsigned short *W0z, *W1z;   // ff0.ws (NT=32,KT=5), ff1.ws (NT=8,KT=17)
};
__launch_bounds__(128)
__global__ void k_node_ff(NodeFFArgs a) {
    __shared__ float ls[16 * 128];
    __shared__ float lv[16 * 48];
    __shared__ float lvh[16 * 96];
    __shared__ float lfv[16 * 96];
    __shared__ alignas(16) unsigned short lA[16 * 544];
    __shared__ float lOut[16 * 512];
    __shared__ float lred[128], lstat[16], lstat2[16], lc[16];
    const int tid = threadIdx.x, lane = tid & 31, wid = tid >> 5;
    const int n0 = blockIdx.x * 16;

    if (tid < 16) lc[tid] = fmaxf(a.CNT[n0 + tid], 1.f);
    __syncthreads();
    for (int i = tid; i < 16 * 128; i += 128) {
        int e = i >> 7, k = i & 127;
        ls[i] = a.S[(size_t)(n0 + e) * 128 + k] + a.DS[(size_t)(n0 + e) * 128 + k] / lc[e];
    }
    for (int i = tid; i < 16 * 48; i += 128) {
        int e = i / 48, j = i - e * 48;
        lv[i] = a.V[(size_t)(n0 + e) * 48 + j] + a.DV[(size_t)(n0 + e) * 48 + j] / lc[e];
    }
    __syncthreads();
    ln16(ls, lred, lstat, lstat2, a.ln0g, a.ln0b, tid);
    rms16(lv, lstat, tid);

    // ff0: vh = wh[32x16] @ v
    for (int i = tid; i < 16 * 96; i += 128) {
        int e = i / 96, j = i - e * 96, h = j / 3, cc = j - h * 3;
        float s = 0.f;
        for (int vv = 0; vv < 16; ++vv) s += a.ff0wh[h * 16 + vv] * lv[e * 48 + vv * 3 + cc];
        lvh[i] = s;
    }
    __syncthreads();
    // A0: cols 0..127 = s (post-ln0), 128..159 = ||vh||
    for (int i = tid; i < 16 * 32; i += 128) {
        int e = i >> 5, h = i & 31;
        float n1 = lvh[e * 96 + h * 3], n2 = lvh[e * 96 + h * 3 + 1], n3 = lvh[e * 96 + h * 3 + 2];
        lA[e * 544 + 128 + h] = f2bf(sqrtf(fmaxf(n1 * n1 + n2 * n2 + n3 * n3, EPSF)));
    }
    for (int i = tid; i < 16 * 128; i += 128) {
        int e = i >> 7, k = i & 127;
        lA[e * 544 + k] = f2bf(ls[i]);
    }
    __syncthreads();
    gemm_tile<5, 544, 32, 1>(lA, a.W0z, lOut, lane, wid);  // fs_raw[16x512]
    __syncthreads();

    // fv = gate(wv[32x32] @ vh)
    for (int i = tid; i < 16 * 32; i += 128) {
        int e = i >> 5, o = i & 31;
        float s0 = 0, s1 = 0, s2 = 0;
        for (int h = 0; h < 32; ++h) {
            float w = a.ff0wv[o * 32 + h];
            s0 += w * lvh[e * 96 + h * 3]; s1 += w * lvh[e * 96 + h * 3 + 1]; s2 += w * lvh[e * 96 + h * 3 + 2];
        }
        float g = sigm(sqrtf(fmaxf(s0 * s0 + s1 * s1 + s2 * s2, EPSF)));
        lfv[e * 96 + o * 3] = s0 * g; lfv[e * 96 + o * 3 + 1] = s1 * g; lfv[e * 96 + o * 3 + 2] = s2 * g;
    }
    __syncthreads();
    // ff1: vh1 = wh[32x32] @ fv
    for (int i = tid; i < 16 * 96; i += 128) {
        int e = i / 96, j = i - e * 96, h = j / 3, cc = j - h * 3;
        float s = 0.f;
        for (int vv = 0; vv < 32; ++vv) s += a.ff1wh[h * 32 + vv] * lfv[e * 96 + vv * 3 + cc];
        lvh[i] = s;
    }
    __syncthreads();
    // A1: cols 0..511 = relu(fs_raw + b), 512..543 = ||vh1||
    for (int i = tid; i < 16 * 32; i += 128) {
        int e = i >> 5, h = i & 31;
        float n1 = lvh[e * 96 + h * 3], n2 = lvh[e * 96 + h * 3 + 1], n3 = lvh[e * 96 + h * 3 + 2];
        lA[e * 544 + 512 + h] = f2bf(sqrtf(fmaxf(n1 * n1 + n2 * n2 + n3 * n3, EPSF)));
    }
    for (int i = tid; i < 16 * 512; i += 128) {
        int e = i >> 9, k = i & 511;
        lA[e * 544 + k] = f2bf(fmaxf(lOut[i] + a.ff0b[k], 0.f));
    }
    __syncthreads();
    gemm_tile<17, 544, 8, 1>(lA, a.W1z, lOut, lane, wid);  // fs2_raw[16x128]
    __syncthreads();

    // residuals: s += fs2 + b, v += wv[16x32] @ vh1
    for (int i = tid; i < 16 * 48; i += 128) {
        int e = i / 48, j = i - e * 48, o = j / 3, cc = j - o * 3;
        float s = 0.f;
        for (int h = 0; h < 32; ++h) s += a.ff1wv[o * 32 + h] * lvh[e * 96 + h * 3 + cc];
        lv[i] += s;
    }
    for (int i = tid; i < 16 * 128; i += 128) {
        int e = i >> 7, k = i & 127;
        ls[i] += lOut[e * 128 + k] + a.ff1b[k];
    }
    __syncthreads();
    ln16(ls, lred, lstat, lstat2, a.ln1g, a.ln1b, tid);
    rms16(lv, lstat, tid);
    for (int i = tid; i < 16 * 128; i += 128) {
        int e = i >> 7, k = i & 127;
        a.S[(size_t)(n0 + e) * 128 + k] = ls[i];
    }
    for (int i = tid; i < 16 * 48; i += 128) {
        int e = i / 48, j = i - e * 48;
        a.V[(size_t)(n0 + e) * 48 + j] = lv[i];
    }
}

// ---------------- output head (WMMA, 16 nodes/block) -----------------------
struct OutArgs {
    const float *S, *V, *lnb, *lng, *wh, *b;
    const unsigned short* Wz;          // Wout.ws (NT=8,KT=5) K=144 pad 160
    float* out;
};
__launch_bounds__(128)
__global__ void k_out(OutArgs a) {
    __shared__ float ls[16 * 128];
    __shared__ float lv[16 * 48];
    __shared__ float lvh[16 * 48];
    __shared__ alignas(16) unsigned short lA[16 * 160];
    __shared__ float lOut[16 * 128];
    __shared__ float lred[128], lstat[16], lstat2[16];
    const int tid = threadIdx.x, lane = tid & 31, wid = tid >> 5;
    const int n0 = blockIdx.x * 16;
    for (int i = tid; i < 16 * 128; i += 128) {
        int e = i >> 7, k = i & 127;
        ls[i] = a.S[(size_t)(n0 + e) * 128 + k];
    }
    for (int i = tid; i < 16 * 48; i += 128) {
        int e = i / 48, j = i - e * 48;
        lv[i] = a.V[(size_t)(n0 + e) * 48 + j];
    }
    __syncthreads();
    ln16(ls, lred, lstat, lstat2, a.lng, a.lnb, tid);
    rms16(lv, lstat, tid);
    for (int i = tid; i < 16 * 48; i += 128) {
        int e = i / 48, j = i - e * 48, h = j / 3, cc = j - h * 3;
        float s = 0.f;
        for (int vv = 0; vv < 16; ++vv) s += a.wh[h * 16 + vv] * lv[e * 48 + vv * 3 + cc];
        lvh[i] = s;
    }
    __syncthreads();
    for (int i = tid; i < 16 * 16; i += 128) {
        int e = i >> 4, h = i & 15;
        float n1 = lvh[e * 48 + h * 3], n2 = lvh[e * 48 + h * 3 + 1], n3 = lvh[e * 48 + h * 3 + 2];
        lA[e * 160 + 128 + h] = f2bf(sqrtf(fmaxf(n1 * n1 + n2 * n2 + n3 * n3, EPSF)));
    }
    for (int i = tid; i < 16 * 128; i += 128) {
        int e = i >> 7, k = i & 127;
        lA[e * 160 + k] = f2bf(ls[i]);
    }
    for (int i = tid; i < 16 * 16; i += 128) lA[(i >> 4) * 160 + 144 + (i & 15)] = 0;
    __syncthreads();
    gemm_tile<5, 160, 8, 1>(lA, a.Wz, lOut, lane, wid);
    __syncthreads();
    for (int i = tid; i < 16 * 128; i += 128) {
        int e = i >> 7, k = i & 127;
        a.out[(size_t)(n0 + e) * 128 + k] = fmaxf(lOut[i] + a.b[k], 0.f);
    }
}

// ---------------- host-side launcher ---------------------------------------
extern "C" void kernel_launch(void* const* d_in, const int* in_sizes, int n_in,
                              void* d_out, int out_size, void* d_ws, size_t ws_size,
                              hipStream_t stream) {
    (void)in_sizes; (void)n_in; (void)out_size; (void)ws_size;
    const int N = NNODE, E = NEDGE;
    auto P = [&](int i) -> const float* { return (const float*)d_in[i]; };
    // params pytree flatten order (dict keys sorted, layers list in order):
    // 0 W_s | 1-4 We_gvp{b,wh,ws,wv} | 5-6 We_ln{b,g} | 7-9 Wout_gvp{b,wh,ws}
    // 10-11 Wout_ln{b,g} | 12-15 Wv_gvp{b,wh,ws,wv} | 16-17 Wv_ln{b,g}
    // 18+24l: ff0{b,wh,ws,wv} ff1{b,wh,ws,wv} ln0{b,g} ln1{b,g}
    //         msg0{b,wh,ws,wv} msg1{...} msg2{...}
    const int* EI  = (const int*)d_in[94];
    const int* seq = (const int*)d_in[95];

    char* base = (char*)d_ws;
    size_t off = 0;
    auto alloc = [&](size_t bytes) -> char* {
        char* p = base + off;
        off = (off + bytes + 255) & ~(size_t)255;
        return p;
    };
    float* S   = (float*)alloc((size_t)N * 128 * 4);
    float* V   = (float*)alloc((size_t)N * 48 * 4);
    float* ES  = (float*)alloc((size_t)E * 32 * 4);
    float* EV  = (float*)alloc((size_t)E * 3 * 4);
    float* dS  = (float*)alloc((size_t)N * 128 * 4);
    float* dV  = (float*)alloc((size_t)N * 48 * 4);
    float* CNT = (float*)alloc((size_t)N * 4);
    unsigned short *W0z[3], *W1z[3], *W2z[3], *ff0Z[3], *ff1Z[3];
    for (int l = 0; l < 3; ++l) {
        W0z[l]  = (unsigned short*)alloc((size_t)8 * 11 * 512 * 2);
        W1z[l]  = (unsigned short*)alloc((size_t)8 * 5 * 512 * 2);
        W2z[l]  = (unsigned short*)alloc((size_t)8 * 5 * 512 * 2);
        ff0Z[l] = (unsigned short*)alloc((size_t)32 * 5 * 512 * 2);
        ff1Z[l] = (unsigned short*)alloc((size_t)8 * 17 * 512 * 2);
    }
    unsigned short* outZ = (unsigned short*)alloc((size_t)8 * 5 * 512 * 2);
    float* WvT = (float*)alloc((size_t)42 * 128 * 4);

    auto grid = [](int total, int tpb) { return dim3((total + tpb - 1) / tpb); };

    // ---- weight preprocessing
    k_transpose<<<grid(128 * 42, 256), 256, 0, stream>>>(P(14), WvT, 128, 42);
    k_swizzle<<<grid(8 * 5 * 512, 256), 256, 0, stream>>>(P(9), outZ, 144, 5, 8);
    for (int l = 0; l < 3; ++l) {
        const int B = 18 + 24 * l;
        k_swizzle<<<grid(8 * 11 * 512, 256), 256, 0, stream>>>(P(B + 14), W0z[l], 321, 11, 8);
        k_swizzle<<<grid(8 * 5 * 512, 256), 256, 0, stream>>>(P(B + 18), W1z[l], 144, 5, 8);
        k_swizzle<<<grid(8 * 5 * 512, 256), 256, 0, stream>>>(P(B + 22), W2z[l], 144, 5, 8);
        k_swizzle<<<grid(32 * 5 * 512, 256), 256, 0, stream>>>(P(B + 2), ff0Z[l], 160, 5, 32);
        k_swizzle<<<grid(8 * 17 * 512, 256), 256, 0, stream>>>(P(B + 6), ff1Z[l], 544, 17, 8);
    }

    // ---- degree counts
    k_zero<<<grid(N, 256), 256, 0, stream>>>(CNT, N);
    k_count<<<grid(E, 256), 256, 0, stream>>>(EI + E, CNT, E);

    // ---- input embeddings
    {
        NodeEmbArgs na;
        na.node_s = P(90); na.node_v = P(91); na.seq = seq; na.Ws = P(0);
        na.lng = P(17); na.lnb = P(16);
        na.wh = P(13); na.wsT = WvT; na.wv = P(15); na.b = P(12);
        na.S = S; na.V = V;
        k_node_embed<<<N, 128, 0, stream>>>(na);
    }
    {
        EdgeEmbArgs ea;
        ea.edge_s = P(92); ea.edge_v = P(93);
        ea.lng = P(6); ea.lnb = P(5);
        ea.wh = P(2); ea.ws = P(3); ea.wv = P(4); ea.b = P(1);
        ea.ES = ES; ea.EV = EV; ea.E = E;
        k_edge_embed<<<grid(E, 256), 256, 0, stream>>>(ea);
    }

    // ---- GNN layers
    for (int l = 0; l < 3; ++l) {
        const int B = 18 + 24 * l;
        k_zero<<<grid(N * 128, 256), 256, 0, stream>>>(dS, N * 128);
        k_zero<<<grid(N * 48, 256), 256, 0, stream>>>(dV, N * 48);
        EdgeArgs ma;
        ma.S = S; ma.V = V; ma.ES = ES; ma.EV = EV;
        ma.src = EI; ma.dst = EI + E; ma.DS = dS; ma.DV = dV;
        ma.W0 = W0z[l]; ma.W1 = W1z[l]; ma.W2 = W2z[l];
        ma.b0 = P(B + 12); ma.wh0 = P(B + 13); ma.wv0 = P(B + 15);
        ma.b1 = P(B + 16); ma.wh1 = P(B + 17); ma.wv1 = P(B + 19);
        ma.b2 = P(B + 20); ma.wh2 = P(B + 21); ma.wv2 = P(B + 23);
        k_edge_msg<<<E / 32, 128, 0, stream>>>(ma);
        NodeFFArgs fa;
        fa.S = S; fa.V = V; fa.DS = dS; fa.DV = dV; fa.CNT = CNT;
        fa.ln0b = P(B + 8); fa.ln0g = P(B + 9); fa.ln1b = P(B + 10); fa.ln1g = P(B + 11);
        fa.ff0b = P(B); fa.ff0wh = P(B + 1); fa.ff0wv = P(B + 3);
        fa.ff1b = P(B + 4); fa.ff1wh = P(B + 5); fa.ff1wv = P(B + 7);
        fa.W0z = ff0Z[l]; fa.W1z = ff1Z[l];
        k_node_ff<<<N / 16, 128, 0, stream>>>(fa);
    }

    // ---- output head
    {
        OutArgs oa;
        oa.S = S; oa.V = V;
        oa.lnb = P(10); oa.lng = P(11);
        oa.wh = P(8); oa.b = P(7); oa.Wz = outZ;
        oa.out = (float*)d_out;
        k_out<<<N / 16, 128, 0, stream>>>(oa);
    }
}
